// VectorQuantizer_58849641889910
// MI455X (gfx1250) — matrix operand order repaired and checked by reading, not compiled, not run on hardware
//
#include <hip/hip_runtime.h>

typedef __attribute__((ext_vector_type(16))) __bf16 v16bf;
typedef __attribute__((ext_vector_type(4)))  __bf16 v4bf;
typedef __attribute__((ext_vector_type(8)))  float  v8f;

#define K_EMB 512
#define DIM   64

// Prep: e2[k] = ||emb_k||^2 (exact f32) into ws[0..511]; zero loss accumulator ws[512].
__global__ __launch_bounds__(256) void vq_prep(const float* __restrict__ emb,
                                               float* __restrict__ ws) {
  int k = blockIdx.x * blockDim.x + threadIdx.x;
  if (k == 0) ws[K_EMB] = 0.0f;
  if (k < K_EMB) {
    const float4* p = (const float4*)(emb + k * DIM);
    float s = 0.f;
    #pragma unroll
    for (int i = 0; i < DIM / 4; ++i) {
      float4 v = p[i];
      s += v.x * v.x + v.y * v.y + v.z * v.z + v.w * v.w;
    }
    ws[k] = s;
  }
}

// Main fused kernel: 16 rows per wave, 8 waves (128 rows) per block.
__global__ __launch_bounds__(256) void vq_main(const float* __restrict__ z,
                                               const float* __restrict__ emb,
                                               const float* __restrict__ e2g,
                                               float* __restrict__ accp,
                                               float* __restrict__ out_zq,
                                               float* __restrict__ out_idx) {
  __shared__ __bf16 s_eb[K_EMB * DIM];  // 64 KB bf16 codebook (B-matrix layout)
  __shared__ float  s_e2[K_EMB];        // 2 KB ||e||^2

  // Stage codebook f32 -> bf16 and e2 into LDS.
  for (int i = threadIdx.x * 4; i < K_EMB * DIM; i += 256 * 4) {
    float4 v = *(const float4*)(emb + i);
    v4bf b;
    b[0] = (__bf16)v.x; b[1] = (__bf16)v.y; b[2] = (__bf16)v.z; b[3] = (__bf16)v.w;
    *(v4bf*)&s_eb[i] = b;
  }
  for (int i = threadIdx.x; i < K_EMB; i += 256) s_e2[i] = e2g[i];
  __syncthreads();

  const int lane = threadIdx.x & 31;
  const int wid  = threadIdx.x >> 5;
  const int r0   = (blockIdx.x * 8 + wid) * 16;  // first of 16 rows for this wave
  const int h    = lane >> 4;                    // wave half
  const int mr   = lane & 15;
  const int koff = h * 16;

  // Build split A fragments: z = z_hi(bf16) + z_lo(bf16 residual), 2 K-chunks of 32.
  const float* zr = z + (r0 + mr) * DIM;
  v16bf ahi[2], alo[2];
  #pragma unroll
  for (int c = 0; c < 2; ++c) {
    #pragma unroll
    for (int g = 0; g < 2; ++g) {
      const float4* p = (const float4*)(zr + 32 * c + koff + 8 * g);
      float4 u0 = p[0], u1 = p[1];
      float x[8] = {u0.x, u0.y, u0.z, u0.w, u1.x, u1.y, u1.z, u1.w};
      #pragma unroll
      for (int t = 0; t < 8; ++t) {
        __bf16 hb = (__bf16)x[t];
        float  rem = x[t] - (float)hb;
        ahi[c][8 * g + t] = hb;
        alo[c][8 * g + t] = (__bf16)rem;
      }
    }
  }

  // Running argmin over score = ||e||^2 - 2*dot(z,e); 8 row-slots per lane.
  float bv[8];
  int   bi[8];
  #pragma unroll
  for (int j = 0; j < 8; ++j) { bv[j] = 3.402823466e38f; bi[j] = 0; }

  // Software-pipelined tile loop: prefetch next tile's B fragments + e2 while
  // the current tile's WMMA chain executes. Wrap index keeps EXEC all-ones.
  const int lb = mr * DIM + h * 16;  // lane's base offset within a tile
  v16bf b0  = *(const v16bf*)&s_eb[lb];
  v16bf b1  = *(const v16bf*)&s_eb[lb + 32];
  float e2v = s_e2[mr];

  #pragma unroll 2
  for (int t = 0; t < K_EMB / 16; ++t) {
    const int tn = (t + 1) & (K_EMB / 16 - 1);
    const int cn = tn * (16 * DIM) + lb;
    v16bf n0  = *(const v16bf*)&s_eb[cn];
    v16bf n1  = *(const v16bf*)&s_eb[cn + 32];
    float ne2 = s_e2[tn * 16 + mr];

    v8f acc = {0.f, 0.f, 0.f, 0.f, 0.f, 0.f, 0.f, 0.f};
    acc = __builtin_amdgcn_wmma_f32_16x16x32_bf16(false, ahi[0], false, b0, (short)0, acc, false, false);
    acc = __builtin_amdgcn_wmma_f32_16x16x32_bf16(false, alo[0], false, b0, (short)0, acc, false, false);
    acc = __builtin_amdgcn_wmma_f32_16x16x32_bf16(false, ahi[1], false, b1, (short)0, acc, false, false);
    acc = __builtin_amdgcn_wmma_f32_16x16x32_bf16(false, alo[1], false, b1, (short)0, acc, false, false);

    const int col = t * 16 + mr;
    #pragma unroll
    for (int j = 0; j < 8; ++j) {
      float s = fmaf(-2.f, acc[j], e2v);
      if (s < bv[j]) { bv[j] = s; bi[j] = col; }
    }
    b0 = n0; b1 = n1; e2v = ne2;
  }

  // Butterfly-reduce argmin across the 16 lanes of each half (first-occurrence ties).
  #pragma unroll
  for (int m = 1; m <= 8; m <<= 1) {
    #pragma unroll
    for (int j = 0; j < 8; ++j) {
      float ov = __shfl_xor(bv[j], m, 32);
      int   oi = __shfl_xor(bi[j], m, 32);
      if (ov < bv[j] || (ov == bv[j] && oi < bi[j])) { bv[j] = ov; bi[j] = oi; }
    }
  }

  if (mr == 0) {  // lanes 0 and 16 hold rows r0..r0+7 and r0+8..r0+15
    #pragma unroll
    for (int j = 0; j < 8; ++j)
      out_idx[r0 + h * 8 + j] = (float)bi[j];
  }

  // Gather z_q = emb[idx] (exact f32), write straight-through output, accumulate MSE.
  float lsum = 0.f;
  #pragma unroll
  for (int m = 0; m < 16; ++m) {
    const int idx = __shfl(bi[m & 7], (m < 8) ? 0 : 16, 32);
    const int row = r0 + m;
    float2 e  = *(const float2*)(emb + idx * DIM + 2 * lane);
    float2 zv = *(const float2*)(z + row * DIM + 2 * lane);
    *(float2*)(out_zq + row * DIM + 2 * lane) = e;
    float d0 = e.x - zv.x, d1 = e.y - zv.y;
    lsum += d0 * d0 + d1 * d1;
  }
  #pragma unroll
  for (int m = 1; m < 32; m <<= 1) lsum += __shfl_xor(lsum, m, 32);
  if (lane == 0) atomicAdd(accp, lsum);
}

__global__ void vq_fin(const float* __restrict__ accp, float* __restrict__ out_loss,
                       float scale) {
  out_loss[0] = accp[0] * scale;
}

extern "C" void kernel_launch(void* const* d_in, const int* in_sizes, int n_in,
                              void* d_out, int out_size, void* d_ws, size_t ws_size,
                              hipStream_t stream) {
  const float* z   = (const float*)d_in[0];
  const float* emb = (const float*)d_in[1];
  const int numel  = in_sizes[0];       // 32*64*64*64 = 8388608
  const int nrows  = numel / DIM;       // 131072

  float* out      = (float*)d_out;
  float* out_zq   = out;                // [numel]
  float* out_loss = out + numel;        // [1]
  float* out_idx  = out + numel + 1;    // [nrows]

  float* ws   = (float*)d_ws;           // ws[0..511]=||e||^2, ws[512]=loss accum
  float* e2g  = ws;
  float* accp = ws + K_EMB;

  vq_prep<<<2, 256, 0, stream>>>(emb, ws);
  vq_main<<<nrows / 128, 256, 0, stream>>>(z, emb, e2g, accp, out_zq, out_idx);
  vq_fin<<<1, 1, 0, stream>>>(accp, out_loss, 1.25f / (float)numel);
}
